// MultiScaleGRU_4904852652068
// MI455X (gfx1250) — matrix operand order repaired and checked by reading, not compiled, not run on hardware
//
#include <hip/hip_runtime.h>
#include <cstddef>

// ---------------------------------------------------------------------------
// MultiScaleGRU for MI455X (gfx1250, wave32, WMMA), f16 operand pipeline with
// async-to-LDS B-panel staging. Dims: L=3, C=4, GH=64, NS=256, NI=128, B=16,
// T=4096. Canonical internal activation layout: row m = t*16 + b, f16.
// ---------------------------------------------------------------------------

typedef __attribute__((ext_vector_type(16))) _Float16 v16h;
typedef __attribute__((ext_vector_type(8)))  _Float16 v8h;
typedef __attribute__((ext_vector_type(8)))  float    v8f;
typedef __attribute__((ext_vector_type(4)))  float    f32x4;

#define L_  3
#define C_  4
#define GH_ 64
#define NS_ 256
#define NI_ 128
#define B_  16
#define T_  4096
#define M_  (B_ * T_)          // 65536 rows

#if __has_builtin(__builtin_amdgcn_global_load_async_to_lds_b128)
#define HAVE_ASYNC_LDS 1
typedef int v4i __attribute__((vector_size(16)));
typedef __attribute__((address_space(1))) v4i as1_v4i;   // global b128 unit
typedef __attribute__((address_space(3))) v4i as3_v4i;   // LDS b128 unit
#else
#define HAVE_ASYNC_LDS 0
#endif

__device__ __forceinline__ void wait_async1() {
#if HAVE_ASYNC_LDS
#if __has_builtin(__builtin_amdgcn_s_wait_asynccnt)
  __builtin_amdgcn_s_wait_asynccnt(1);
#else
  asm volatile("s_wait_asynccnt 0x1" ::: "memory");
#endif
#endif
}
__device__ __forceinline__ void wait_async0() {
#if HAVE_ASYNC_LDS
#if __has_builtin(__builtin_amdgcn_s_wait_asynccnt)
  __builtin_amdgcn_s_wait_asynccnt(0);
#else
  asm volatile("s_wait_asynccnt 0x0" ::: "memory");
#endif
#endif
}

// Stage one 64x32 f16 B panel chunk into LDS: 256 threads x 16B = 4KB.
__device__ __forceinline__ void stage_b(const _Float16* __restrict__ Wchunk,
                                        _Float16* __restrict__ lbase,
                                        int tid, int Kstride) {
  const _Float16* gsrc = Wchunk + (size_t)(tid >> 2) * Kstride + (tid & 3) * 8;
  _Float16* ldst = lbase + tid * 8;
#if HAVE_ASYNC_LDS
  __builtin_amdgcn_global_load_async_to_lds_b128((as1_v4i*)gsrc, (as3_v4i*)ldst, 0, 0);
#else
  *(v8h*)ldst = *(const v8h*)gsrc;
#endif
}

__device__ __forceinline__ v8f vzero8() {
  v8f z = {0.f, 0.f, 0.f, 0.f, 0.f, 0.f, 0.f, 0.f};
  return z;
}
__device__ __forceinline__ float sigmoid_fast(float x) {
  return 1.0f / (1.0f + __expf(-x));
}
__device__ __forceinline__ float tanh_fast(float x) {
  return 2.0f / (1.0f + __expf(-2.0f * x)) - 1.0f;
}
__device__ __forceinline__ v16h join16(v8h lo, v8h hi) {
  v16h a;
#pragma unroll
  for (int i = 0; i < 8; ++i) { a[i] = lo[i]; a[8 + i] = hi[i]; }
  return a;
}

// ---------------------------------------------------------------------------
// Elementwise f32 -> f16 pack (weights). n8 = count/8.
// ---------------------------------------------------------------------------
__global__ __launch_bounds__(256)
void f32_to_f16_kernel(const float* __restrict__ src, _Float16* __restrict__ dst,
                       int n8) {
  const int i = blockIdx.x * 256 + threadIdx.x;
  if (i >= n8) return;
  const float* s = src + (size_t)i * 8;
  f32x4 a = *(const f32x4*)s;
  f32x4 b = *(const f32x4*)(s + 4);
  v8h o;
#pragma unroll
  for (int j = 0; j < 4; ++j) { o[j] = (_Float16)a[j]; o[4 + j] = (_Float16)b[j]; }
  *(v8h*)(dst + (size_t)i * 8) = o;
}

// Encoder input pack: (B,T,NI) f32 -> canonical (t*16+b, NI) f16.
__global__ __launch_bounds__(256)
void pack_inputs_kernel(const float* __restrict__ src, _Float16* __restrict__ dst) {
  const int i = blockIdx.x * 256 + threadIdx.x;   // 0 .. M_*16-1
  const int m = i >> 4;
  const int kk = (i & 15) * 8;
  const int t = m >> 4, b = m & 15;
  const float* s = src + ((size_t)b * T_ + t) * NI_ + kk;
  f32x4 a = *(const f32x4*)s;
  f32x4 c = *(const f32x4*)(s + 4);
  v8h o;
#pragma unroll
  for (int j = 0; j < 4; ++j) { o[j] = (_Float16)a[j]; o[4 + j] = (_Float16)c[j]; }
  *(v8h*)(dst + (size_t)m * NI_ + kk) = o;
}

// ---------------------------------------------------------------------------
// GEMM: acc = Xh @ Wh.T (+bias, opt tanh). f16 operands. N = 256, M = 65536.
// Block 256 thr (8 waves); wave -> 32 rows x 64 cols (2x4 tiles); block ->
// 256x64. Grid (256, 4). B panel (64x32 f16) staged to LDS with async copies,
// double-buffered over the K loop. Per chunk/wave: 4 global + 8 ds + 8 wmma.
// ---------------------------------------------------------------------------
template<int K, bool TANH, bool OUT16, bool OUT32BT>
__global__ __launch_bounds__(256)
void gemm16_kernel(const _Float16* __restrict__ Xh, const _Float16* __restrict__ Wh,
                   const float* __restrict__ bias, _Float16* __restrict__ Y16,
                   float* __restrict__ Y32) {
  constexpr int NK = K / 32;
  const int tid  = threadIdx.x;
  const int wv   = tid >> 5;
  const int lane = tid & 31;
  const int r    = lane & 15;
  const int hf   = lane >> 4;
  const int m0   = blockIdx.x * 256 + wv * 32;
  const int nb   = blockIdx.y * 64;

  __shared__ __align__(16) _Float16 Bp[2][64 * 32];

  const _Float16* a0row = Xh + (size_t)(m0 + r) * K;
  const _Float16* a1row = Xh + (size_t)(m0 + 16 + r) * K;
  const _Float16* Wnb   = Wh + (size_t)nb * K;

  v8f acc[2][4];
#pragma unroll
  for (int rt = 0; rt < 2; ++rt)
#pragma unroll
    for (int tt = 0; tt < 4; ++tt) acc[rt][tt] = vzero8();

  stage_b(Wnb, &Bp[0][0], tid, K);                 // chunk 0 -> buf 0

#pragma unroll
  for (int kc = 0; kc < NK; ++kc) {
    if (kc + 1 < NK) {
      if (kc >= 1) __syncthreads();                // buf (kc+1)&1 free to overwrite
      stage_b(Wnb + (kc + 1) * 32, &Bp[(kc + 1) & 1][0], tid, K);
      wait_async1();                               // chunk kc landed
    } else {
      wait_async0();
    }
    __syncthreads();                               // chunk kc visible to all

    const v16h af0 = join16(*(const v8h*)(a0row + kc * 32 + hf * 8),
                            *(const v8h*)(a0row + kc * 32 + 16 + hf * 8));
    const v16h af1 = join16(*(const v8h*)(a1row + kc * 32 + hf * 8),
                            *(const v8h*)(a1row + kc * 32 + 16 + hf * 8));
    const _Float16* bp = &Bp[kc & 1][0];
#pragma unroll
    for (int tt = 0; tt < 4; ++tt) {
      const _Float16* brow = bp + (tt * 16 + r) * 32 + hf * 8;
      const v16h bf = join16(*(const v8h*)brow, *(const v8h*)(brow + 16));
      acc[0][tt] = __builtin_amdgcn_wmma_f32_16x16x32_f16(
          false, af0, false, bf, (short)0, acc[0][tt], false, false);
      acc[1][tt] = __builtin_amdgcn_wmma_f32_16x16x32_f16(
          false, af1, false, bf, (short)0, acc[1][tt], false, false);
    }
  }

#pragma unroll
  for (int rt = 0; rt < 2; ++rt) {
#pragma unroll
    for (int tt = 0; tt < 4; ++tt) {
      const int col = nb + tt * 16 + r;
      const float bv = bias[col];
#pragma unroll
      for (int rr = 0; rr < 8; ++rr) {
        const int mr = m0 + rt * 16 + (hf ? 8 : 0) + rr;
        float v = acc[rt][tt][rr] + bv;
        if (TANH) v = tanh_fast(v);
        if (OUT16) Y16[(size_t)mr * NS_ + col] = (_Float16)v;
        if (OUT32BT) {
          const int t = mr >> 4, b = mr & 15;
          Y32[((size_t)b * T_ + t) * NS_ + col] = v;
        }
      }
    }
  }
}

// ---------------------------------------------------------------------------
// Input-gate precompute: ig = (x @ Wih.T) * (1/s) + b; f16 operands, f32
// fragment-layout output igbuf[c][t][tile(12)][lane(32)][elem(8)].
// Same 2x4-tile / async-staged structure. Grid (256, 3, C_), block 256.
// ---------------------------------------------------------------------------
__global__ __launch_bounds__(256)
void ig_gemm_kernel(const _Float16* __restrict__ Xh, const _Float16* __restrict__ Wih,
                    const float* __restrict__ bvec, const float* __restrict__ log_s,
                    float* __restrict__ igbuf) {
  constexpr int NK = NS_ / 32;
  const int c    = blockIdx.z;
  const int tid  = threadIdx.x;
  const int wv   = tid >> 5;
  const int lane = tid & 31;
  const int r    = lane & 15;
  const int hf   = lane >> 4;
  const int m0   = blockIdx.x * 256 + wv * 32;
  const int nb   = blockIdx.y * 64;

  __shared__ __align__(16) _Float16 Bp[2][64 * 32];

  const float inv_s = __expf(-log_s[c]);           // applied post-accumulation
  const _Float16* Wnb = Wih + (size_t)c * 192 * NS_ + (size_t)nb * NS_;
  const float* bb     = bvec + c * 192;
  const _Float16* a0row = Xh + (size_t)(m0 + r) * NS_;
  const _Float16* a1row = Xh + (size_t)(m0 + 16 + r) * NS_;

  v8f acc[2][4];
#pragma unroll
  for (int rt = 0; rt < 2; ++rt)
#pragma unroll
    for (int tt = 0; tt < 4; ++tt) acc[rt][tt] = vzero8();

  stage_b(Wnb, &Bp[0][0], tid, NS_);

#pragma unroll
  for (int kc = 0; kc < NK; ++kc) {
    if (kc + 1 < NK) {
      if (kc >= 1) __syncthreads();
      stage_b(Wnb + (kc + 1) * 32, &Bp[(kc + 1) & 1][0], tid, NS_);
      wait_async1();
    } else {
      wait_async0();
    }
    __syncthreads();

    const v16h af0 = join16(*(const v8h*)(a0row + kc * 32 + hf * 8),
                            *(const v8h*)(a0row + kc * 32 + 16 + hf * 8));
    const v16h af1 = join16(*(const v8h*)(a1row + kc * 32 + hf * 8),
                            *(const v8h*)(a1row + kc * 32 + 16 + hf * 8));
    const _Float16* bp = &Bp[kc & 1][0];
#pragma unroll
    for (int tt = 0; tt < 4; ++tt) {
      const _Float16* brow = bp + (tt * 16 + r) * 32 + hf * 8;
      const v16h bf = join16(*(const v8h*)brow, *(const v8h*)(brow + 16));
      acc[0][tt] = __builtin_amdgcn_wmma_f32_16x16x32_f16(
          false, af0, false, bf, (short)0, acc[0][tt], false, false);
      acc[1][tt] = __builtin_amdgcn_wmma_f32_16x16x32_f16(
          false, af1, false, bf, (short)0, acc[1][tt], false, false);
    }
  }

  const int t0 = m0 >> 4;                          // tile rt covers time t0+rt
#pragma unroll
  for (int rt = 0; rt < 2; ++rt) {
#pragma unroll
    for (int tt = 0; tt < 4; ++tt) {
      const int j = (nb >> 4) + tt;                // global tile index 0..11
      const float bv = bb[j * 16 + r];
      float* op = igbuf + ((((size_t)c * T_ + (t0 + rt)) * 12 + j) * 32 + lane) * 8;
      f32x4 lo, hi;
#pragma unroll
      for (int i = 0; i < 4; ++i) {
        lo[i] = acc[rt][tt][i]     * inv_s + bv;
        hi[i] = acc[rt][tt][4 + i] * inv_s + bv;
      }
      *(f32x4*)(op + 0) = lo;
      *(f32x4*)(op + 4) = hi;
    }
  }
}

// ---------------------------------------------------------------------------
// Sequential GRU scan, one channel per block, 4 waves. Wave w owns hidden
// slice [16w,16w+16): r/z/n tiles wave-local; Whh B-fragments resident in
// VGPRs; h double-buffered in LDS as f16 in WMMA-A layout.
// Per step: 4 ds_load_b128 + 6 global b128 (ig) + 6 v_wmma + gates.
// ---------------------------------------------------------------------------
__global__ __launch_bounds__(128)
void gru_scan_kernel(const float* __restrict__ igbuf, const float* __restrict__ Whh,
                     const float* __restrict__ bn, const float* __restrict__ log_s,
                     const float* __restrict__ h0, float* __restrict__ ynew,
                     _Float16* __restrict__ xcat) {
  const int c    = blockIdx.x;
  const int tid  = threadIdx.x;
  const int w    = tid >> 5;
  const int lane = tid & 31;
  const int r    = lane & 15;
  const int hf   = lane >> 4;
  const int hid  = w * 16 + r;

  const float inv_s = __expf(-log_s[c]);
  const float bnv   = bn[c * GH_ + hid];

  v16h Bfr[3][2];
#pragma unroll
  for (int g = 0; g < 3; ++g) {
#pragma unroll
    for (int kc = 0; kc < 2; ++kc) {
      const float* wr = Whh + ((size_t)c * 192 + g * GH_ + hid) * GH_ + kc * 32 + hf * 8;
      v16h bf;
#pragma unroll
      for (int i = 0; i < 16; ++i) bf[i] = (_Float16)wr[i < 8 ? i : i + 8];
      Bfr[g][kc] = bf;
    }
  }

  v8f hreg;
#pragma unroll
  for (int rr = 0; rr < 8; ++rr) {
    const int b = (hf ? 8 : 0) + rr;
    hreg[rr] = h0[((size_t)c * B_ + b) * GH_ + hid];
  }

  __shared__ __align__(16) _Float16 hlds[2][B_][GH_];
#pragma unroll
  for (int rr = 0; rr < 8; ++rr) {
    const int b = (hf ? 8 : 0) + rr;
    hlds[0][b][hid] = (_Float16)hreg[rr];
  }
  __syncthreads();

  const float* igc = igbuf + (size_t)c * T_ * 12 * 256;
  int p = 0;
  for (int t = 0; t < T_; ++t) {
    v16h af[2];
#pragma unroll
    for (int kc = 0; kc < 2; ++kc) {
      const _Float16* hp = &hlds[p][r][0];
      v8h lo = *(const v8h*)(hp + kc * 32 + hf * 8);
      v8h hi = *(const v8h*)(hp + kc * 32 + 16 + hf * 8);
      af[kc] = join16(lo, hi);
    }

    const int tp = (t + 8 < T_) ? (t + 8) : (T_ - 1);
    v8f igv[3];
#pragma unroll
    for (int g = 0; g < 3; ++g) {
      const float* ip = igc + ((size_t)t * 12 + (g * 4 + w)) * 256 + lane * 8;
      __builtin_prefetch(igc + ((size_t)tp * 12 + (g * 4 + w)) * 256 + lane * 8, 0, 0);
      f32x4 lo = *(const f32x4*)(ip + 0);
      f32x4 hi = *(const f32x4*)(ip + 4);
      v8f v;
#pragma unroll
      for (int i = 0; i < 4; ++i) { v[i] = lo[i]; v[4 + i] = hi[i]; }
      igv[g] = v;
    }

    v8f accg[3];
#pragma unroll
    for (int g = 0; g < 3; ++g) {
      v8f a = vzero8();
      a = __builtin_amdgcn_wmma_f32_16x16x32_f16(false, af[0], false, Bfr[g][0],
                                                 (short)0, a, false, false);
      a = __builtin_amdgcn_wmma_f32_16x16x32_f16(false, af[1], false, Bfr[g][1],
                                                 (short)0, a, false, false);
      accg[g] = a;
    }

    v8f ynext;
#pragma unroll
    for (int rr = 0; rr < 8; ++rr) {
      const float rg = sigmoid_fast(igv[0][rr] + accg[0][rr]);
      const float zg = sigmoid_fast(igv[1][rr] + accg[1][rr]);
      const float ng = tanh_fast(igv[2][rr] + rg * (accg[2][rr] + bnv));
      const float ho = hreg[rr];
      const float hn = ng + zg * (ho - ng);        // eqx GRU update
      ynext[rr] = (hn - ho) * inv_s + ho;          // ScaleGRU rescale; carry=y
    }
    hreg = ynext;

#pragma unroll
    for (int rr = 0; rr < 8; ++rr) {
      const int b = (hf ? 8 : 0) + rr;
      const float y = ynext[rr];
      ynew[(((size_t)c * B_ + b) * T_ + t) * GH_ + hid] = y;
      xcat[((size_t)t * B_ + b) * NS_ + c * GH_ + hid] = (_Float16)y;
      hlds[p ^ 1][b][hid] = (_Float16)y;
    }
    __syncthreads();
    p ^= 1;
  }
}

// ---------------------------------------------------------------------------
// Host orchestration (all on `stream`, graph-capture safe).
// ws layout: igbuf f32 (201.3MB) | actA f16 (32MB) | actB f16 (32MB) |
//            in16 f16 (16MB) | packed weights f16 (~2.2MB)
// ---------------------------------------------------------------------------
extern "C" void kernel_launch(void* const* d_in, const int* in_sizes, int n_in,
                              void* d_out, int out_size, void* d_ws, size_t ws_size,
                              hipStream_t stream) {
  (void)in_sizes; (void)n_in; (void)out_size; (void)ws_size;

  const float* inputs  = (const float*)d_in[0];
  const float* h0      = (const float*)d_in[1];
  // d_in[2] = yinit_guess (unused by the reference computation)
  const float* enc_W1  = (const float*)d_in[3];
  const float* enc_b1  = (const float*)d_in[4];
  const float* enc_W2  = (const float*)d_in[5];
  const float* enc_b2  = (const float*)d_in[6];
  const float* gru_Wih = (const float*)d_in[7];
  const float* gru_Whh = (const float*)d_in[8];
  const float* gru_b   = (const float*)d_in[9];
  const float* gru_bn  = (const float*)d_in[10];
  const float* log_s   = (const float*)d_in[11];
  const float* mlp_W1  = (const float*)d_in[12];
  const float* mlp_b1  = (const float*)d_in[13];
  const float* mlp_W2  = (const float*)d_in[14];
  const float* mlp_b2  = (const float*)d_in[15];

  float* out  = (float*)d_out;                     // (B,T,NS)
  float* ynew = out + (size_t)B_ * T_ * NS_;       // (L,C,B,T,GH)

  char* wsb = (char*)d_ws;
  float*     igbuf = (float*)wsb;
  _Float16*  actA  = (_Float16*)(wsb + 201326592u);
  _Float16*  actB  = (_Float16*)(wsb + 201326592u + 33554432u);
  _Float16*  in16  = (_Float16*)(wsb + 201326592u + 2u * 33554432u);
  _Float16*  wEnc1 = (_Float16*)(wsb + 201326592u + 2u * 33554432u + 16777216u);
  _Float16*  wEnc2 = wEnc1 + 256 * 128;
  _Float16*  wMlp1 = wEnc2 + 256 * 256;
  _Float16*  wMlp2 = wMlp1 + 3 * 256 * 256;
  _Float16*  wIh   = wMlp2 + 3 * 256 * 256;

  auto cvt = [&](const float* s, _Float16* d, int n) {
    const int n8 = n / 8;
    f32_to_f16_kernel<<<(n8 + 255) / 256, 256, 0, stream>>>(s, d, n8);
  };
  cvt(enc_W1, wEnc1, 256 * 128);
  cvt(enc_W2, wEnc2, 256 * 256);
  cvt(mlp_W1, wMlp1, 3 * 256 * 256);
  cvt(mlp_W2, wMlp2, 3 * 256 * 256);
  cvt(gru_Wih, wIh, L_ * C_ * 3 * GH_ * NS_);
  pack_inputs_kernel<<<(M_ * 16) / 256, 256, 0, stream>>>(inputs, in16);

  const dim3 gblk(256);
  const dim3 ggrd(M_ / 256, NS_ / 64, 1);          // (256, 4)

  gemm16_kernel<NI_, true,  true, false><<<ggrd, gblk, 0, stream>>>(
      in16, wEnc1, enc_b1, actA, (float*)nullptr);
  gemm16_kernel<NS_, false, true, false><<<ggrd, gblk, 0, stream>>>(
      actA, wEnc2, enc_b2, actB, (float*)nullptr);

  _Float16* X = actB;
  _Float16* S = actA;
  for (int l = 0; l < L_; ++l) {
    ig_gemm_kernel<<<dim3(M_ / 256, 3, C_), gblk, 0, stream>>>(
        X, wIh + (size_t)l * C_ * 3 * GH_ * NS_,
        gru_b + (size_t)l * C_ * 3 * GH_,
        log_s + (size_t)l * C_, igbuf);

    gru_scan_kernel<<<dim3(C_), dim3(128), 0, stream>>>(
        igbuf, gru_Whh + (size_t)l * C_ * 3 * GH_ * GH_,
        gru_bn + (size_t)l * C_ * GH_,
        log_s + (size_t)l * C_,
        h0 + (size_t)l * C_ * B_ * GH_,
        ynew + (size_t)l * C_ * B_ * T_ * GH_,
        S);

    gemm16_kernel<NS_, true,  true, false><<<ggrd, gblk, 0, stream>>>(
        S, wMlp1 + (size_t)l * NS_ * NS_, mlp_b1 + (size_t)l * NS_, X, (float*)nullptr);
    gemm16_kernel<NS_, false, true, false><<<ggrd, gblk, 0, stream>>>(
        X, wMlp2 + (size_t)l * NS_ * NS_, mlp_b2 + (size_t)l * NS_, S, (float*)nullptr);
    _Float16* tmp = X; X = S; S = tmp;
  }

  gemm16_kernel<NS_, true,  true,  false><<<ggrd, gblk, 0, stream>>>(
      X, wMlp1 + (size_t)(L_ - 1) * NS_ * NS_, mlp_b1 + (size_t)(L_ - 1) * NS_,
      S, (float*)nullptr);
  gemm16_kernel<NS_, false, false, true><<<ggrd, gblk, 0, stream>>>(
      S, wMlp2 + (size_t)(L_ - 1) * NS_ * NS_, mlp_b2 + (size_t)(L_ - 1) * NS_,
      (_Float16*)nullptr, out);
}
